// DualHeterogeneousTransformer_24670292148411
// MI455X (gfx1250) — matrix operand order repaired and checked by reading, not compiled, not run on hardware
//
#include <hip/hip_runtime.h>
#include <stdint.h>

// ---------------------------------------------------------------------------
// DualHeterogeneousTransformer forward, MI455X (gfx1250) version.
// Only out[:,0] is returned -> only Q row 0 needed; relation K/V split into
// per-batch + per-position terms. Heavy work = entity K/V projection
// (131072x512x512 x2, bf16 WMMA, fp32 accum), fused with attention per batch.
// Round-2 changes: (a) batch all 4 A-fragment LDS loads per k-tile and issue
// 4 back-to-back v_wmma (avoid per-wmma s_wait_dscnt serialization);
// (b) non-temporal loads for the 256MB once-read qee stream so the 2.5MB of
// bf16 weights stay resident in the 192MB L2 across all 2048 workgroups.
// ---------------------------------------------------------------------------

#define D    512
#define LSEQ 64
#define NB   2048

typedef __bf16 v16bf __attribute__((ext_vector_type(16)));
typedef float  v8f   __attribute__((ext_vector_type(8)));
typedef float  f32x4 __attribute__((ext_vector_type(4)));

struct __align__(16) U128 { unsigned x, y, z, w; };
struct __align__(8)  U64p { unsigned lo, hi; };

union FragBF { U128 u[2]; v16bf v; };

static __device__ __forceinline__ unsigned short f2bf(float f) {
  union { float f; unsigned u; } v; v.f = f;
  unsigned r = v.u + 0x7FFFu + ((v.u >> 16) & 1u);   // round-to-nearest-even
  return (unsigned short)(r >> 16);
}
static __device__ __forceinline__ float bf2f(unsigned short h) {
  union { unsigned u; float f; } v; v.u = ((unsigned)h) << 16;
  return v.f;
}
static __device__ __forceinline__ U64p pack4(float x, float y, float z, float w) {
  U64p pk;
  pk.lo = ((unsigned)f2bf(y) << 16) | f2bf(x);
  pk.hi = ((unsigned)f2bf(w) << 16) | f2bf(z);
  return pk;
}

// --------------------------- kernel A1: weights -> bf16 ---------------------
__global__ __launch_bounds__(256) void k_convert_weights(
    const float* __restrict__ Wq, const float* __restrict__ Wk,
    const float* __restrict__ Wv, const float* __restrict__ Wrk,
    const float* __restrict__ Wrv, unsigned short* __restrict__ out) {
  unsigned id = blockIdx.x * 256u + threadIdx.x;
  if (id >= 5u * 262144u) return;
  unsigned w = id >> 18, off = id & 0x3FFFFu;
  const float* src = (w == 0) ? Wq : (w == 1) ? Wk : (w == 2) ? Wv
                   : (w == 3) ? Wrk : Wrv;
  out[id] = f2bf(src[off]);
}

// ------------------- kernel A2: tiny fp32 precomputes -----------------------
// posr_k[l][n] = pos_r[l] . Wr_k[n] + br_k[n]      (64x512); posr_v likewise;
// kmask/vmask = (mask_emb+pos_e[64]) projections (1x512 each)
__global__ __launch_bounds__(256) void k_small_precompute(
    const float* __restrict__ mask_emb, const float* __restrict__ pos_e,
    const float* __restrict__ pos_r,
    const float* __restrict__ We_k, const float* __restrict__ be_k,
    const float* __restrict__ We_v, const float* __restrict__ be_v,
    const float* __restrict__ Wr_k, const float* __restrict__ br_k,
    const float* __restrict__ Wr_v, const float* __restrict__ br_v,
    float* __restrict__ posr_k, float* __restrict__ posr_v,
    float* __restrict__ kmask, float* __restrict__ vmask) {
  unsigned id = blockIdx.x * 256u + threadIdx.x;
  if (id < 32768u) {
    unsigned l = id >> 9, n = id & 511u;
    float acc = br_k[n];
    for (int c = 0; c < D; ++c) acc = fmaf(pos_r[l * D + c], Wr_k[n * D + c], acc);
    posr_k[id] = acc;
  } else if (id < 65536u) {
    unsigned i2 = id - 32768u, l = i2 >> 9, n = i2 & 511u;
    float acc = br_v[n];
    for (int c = 0; c < D; ++c) acc = fmaf(pos_r[l * D + c], Wr_v[n * D + c], acc);
    posr_v[i2] = acc;
  } else if (id < 66048u) {
    unsigned n = id - 65536u;
    float acc = be_k[n];
    for (int c = 0; c < D; ++c)
      acc = fmaf(mask_emb[c] + pos_e[64 * D + c], We_k[n * D + c], acc);
    kmask[n] = acc;
  } else if (id < 66560u) {
    unsigned n = id - 66048u;
    float acc = be_v[n];
    for (int c = 0; c < D; ++c)
      acc = fmaf(mask_emb[c] + pos_e[64 * D + c], We_v[n * D + c], acc);
    vmask[n] = acc;
  }
}

// ----------- kernel B: per-batch single-row projections via WMMA ------------
// qrow = (qee[:,0]+pos_e[0]) @ We_q^T + be_q ; relK = relation @ Wr_k^T ;
// relV = relation @ Wr_v^T   (all 2048x512, bf16 WMMA, fp32 out)
__global__ __launch_bounds__(256) void k_row_gemms(
    const float* __restrict__ qee, const float* __restrict__ rel,
    const float* __restrict__ pos_e,
    const unsigned short* __restrict__ WqBf,
    const unsigned short* __restrict__ WrkBf,
    const unsigned short* __restrict__ WrvBf,
    const float* __restrict__ be_q,
    float* __restrict__ qrow, float* __restrict__ relK, float* __restrict__ relV) {
  extern __shared__ unsigned short sm[];
  unsigned short* sQ = sm;           // 64x512 bf16
  unsigned short* sR = sm + 32768;   // 64x512 bf16
  const int t = threadIdx.x;
  const int wg = blockIdx.x;         // rows wg*64 .. wg*64+63

  for (int i = t; i < 8192; i += 256) {         // float4 quads, 64x512
    int row = i >> 7, c4 = (i & 127) * 4;
    int b = wg * 64 + row;
    f32x4 a = *(const f32x4*)(qee + (size_t)b * 64 * D + c4);   // qee[b,0,:]
    f32x4 p = *(const f32x4*)(pos_e + c4);                      // pos_e[0]
    *(U64p*)(sQ + row * D + c4) = pack4(a.x + p.x, a.y + p.y, a.z + p.z, a.w + p.w);
    f32x4 r = *(const f32x4*)(rel + (size_t)b * D + c4);
    *(U64p*)(sR + row * D + c4) = pack4(r.x, r.y, r.z, r.w);
  }
  __syncthreads();

  const int lane = t & 31, wave = t >> 5;
  const int lm = lane & 15, lh = lane >> 4;
  const v8f vz = {0.f, 0.f, 0.f, 0.f, 0.f, 0.f, 0.f, 0.f};

  for (int j = wave; j < 96; j += 8) {          // (matrix, ntile) jobs
    int mat = j >> 5;                           // 0:q 1:relK 2:relV
    int nt = j & 31;
    const unsigned short* A = (mat == 0) ? sQ : sR;
    const unsigned short* W = (mat == 0) ? WqBf : (mat == 1) ? WrkBf : WrvBf;
    v8f acc[4];
#pragma unroll
    for (int mt = 0; mt < 4; ++mt) acc[mt] = vz;
    for (int kt = 0; kt < 16; ++kt) {
      FragBF bfr;
      const unsigned short* bp = W + ((size_t)(nt * 16 + lm) * D + kt * 32 + lh * 8);
      bfr.u[0] = *(const U128*)bp;
      bfr.u[1] = *(const U128*)(bp + 16);
      FragBF afr[4];                  // burst all LDS loads, single wait
#pragma unroll
      for (int mt = 0; mt < 4; ++mt) {
        const unsigned short* ap = A + (mt * 16 + lm) * D + kt * 32 + lh * 8;
        afr[mt].u[0] = *(const U128*)ap;
        afr[mt].u[1] = *(const U128*)(ap + 16);
      }
#pragma unroll
      for (int mt = 0; mt < 4; ++mt)  // 4 back-to-back WMMAs, no hazards
        acc[mt] = __builtin_amdgcn_wmma_f32_16x16x32_bf16(
            false, afr[mt].v, false, bfr.v, (short)0, acc[mt], false, false);
    }
    float bias = (mat == 0) ? be_q[nt * 16 + lm] : 0.0f;
    float* Out = (mat == 0) ? qrow : (mat == 1) ? relK : relV;
#pragma unroll
    for (int mt = 0; mt < 4; ++mt)
#pragma unroll
      for (int i = 0; i < 8; ++i) {
        int row = wg * 64 + mt * 16 + lh * 8 + i;
        Out[(size_t)row * D + nt * 16 + lm] = acc[mt][i] + bias;
      }
  }
}

// -------- kernel C: fused entity K/V projection + attention per batch -------
__global__ __launch_bounds__(256) void k_fused_attn(
    const float* __restrict__ qee, const float* __restrict__ pos_e,
    const unsigned short* __restrict__ WkBf, const unsigned short* __restrict__ WvBf,
    const float* __restrict__ be_k, const float* __restrict__ be_v,
    const float* __restrict__ qrow, const float* __restrict__ relK,
    const float* __restrict__ relV,
    const float* __restrict__ posr_k, const float* __restrict__ posr_v,
    const float* __restrict__ kmask, const float* __restrict__ vmask,
    float* __restrict__ out) {
  extern __shared__ unsigned short sm[];
  unsigned short* sEnt = sm;            // 64x512 bf16 A tile     (64 KB)
  unsigned short* sK   = sm + 32768;    // 64x512 bf16 entity K   (64 KB)
  unsigned short* sV   = sm + 65536;    // 64x512 bf16 entity V   (64 KB)
  float* sc = (float*)(sm + 98304);     // 129 raw scores
  float* sp = sc + 132;                 // 129 exp(scores)

  const int t = threadIdx.x;
  const int b = blockIdx.x;

  // Stage A = qee[b] + pos_e[:64] in bf16. qee is read exactly once across
  // the whole dispatch -> non-temporal so weights stay hot in L2.
  for (int i = t; i < 8192; i += 256) {
    int row = i >> 7, c4 = (i & 127) * 4;
    f32x4 a = __builtin_nontemporal_load(
        (const f32x4*)(qee + ((size_t)b * 64 + row) * D + c4));
    f32x4 p = *(const f32x4*)(pos_e + row * D + c4);
    *(U64p*)(sEnt + row * D + c4) = pack4(a.x + p.x, a.y + p.y, a.z + p.z, a.w + p.w);
  }
  __syncthreads();

  const int lane = t & 31, wave = t >> 5;
  const int lm = lane & 15, lh = lane >> 4;
  const v8f vz = {0.f, 0.f, 0.f, 0.f, 0.f, 0.f, 0.f, 0.f};

  // Entity K and V: 2 matrices x 32 ntiles = 64 jobs, 8 per wave.
  // 4096 v_wmma_f32_16x16x32_bf16 per workgroup (512 per wave).
  for (int j = wave; j < 64; j += 8) {
    int mat = j >> 5;                      // 0: K, 1: V
    int nt = j & 31;
    const unsigned short* W = mat ? WvBf : WkBf;
    const float* bias = mat ? be_v : be_k;
    unsigned short* dst = mat ? sV : sK;
    v8f acc[4];
#pragma unroll
    for (int mt = 0; mt < 4; ++mt) acc[mt] = vz;
    for (int kt = 0; kt < 16; ++kt) {
      FragBF bfr;                          // B[k][n] = W[n][k], streamed from L2
      const unsigned short* bp = W + ((size_t)(nt * 16 + lm) * D + kt * 32 + lh * 8);
      bfr.u[0] = *(const U128*)bp;
      bfr.u[1] = *(const U128*)(bp + 16);
      FragBF afr[4];                       // burst all LDS loads, single wait
#pragma unroll
      for (int mt = 0; mt < 4; ++mt) {
        const unsigned short* ap = sEnt + (mt * 16 + lm) * D + kt * 32 + lh * 8;
        afr[mt].u[0] = *(const U128*)ap;
        afr[mt].u[1] = *(const U128*)(ap + 16);
      }
#pragma unroll
      for (int mt = 0; mt < 4; ++mt)       // 4 back-to-back WMMAs
        acc[mt] = __builtin_amdgcn_wmma_f32_16x16x32_bf16(
            false, afr[mt].v, false, bfr.v, (short)0, acc[mt], false, false);
    }
    float bv = bias[nt * 16 + lm];
#pragma unroll
    for (int mt = 0; mt < 4; ++mt)
#pragma unroll
      for (int i = 0; i < 8; ++i)
        dst[(mt * 16 + lh * 8 + i) * D + nt * 16 + lm] = f2bf(acc[mt][i] + bv);
  }
  __syncthreads();

  // Scores over 129 keys: one key per wave iteration, 16 dims per lane.
  const float* qv = qrow + (size_t)b * D;
  for (int l = wave; l < 129; l += 8) {
    float partial = 0.f;
    int d0 = lane * 16;
    if (l < 64) {
      for (int c = 0; c < 16; ++c)
        partial = fmaf(qv[d0 + c], bf2f(sK[l * D + d0 + c]), partial);
    } else if (l == 64) {
      for (int c = 0; c < 16; ++c)
        partial = fmaf(qv[d0 + c], kmask[d0 + c], partial);
    } else {
      int jr = l - 65;
      const float* rk = relK + (size_t)b * D;
      const float* pk = posr_k + jr * D;
      for (int c = 0; c < 16; ++c)
        partial = fmaf(qv[d0 + c], rk[d0 + c] + pk[d0 + c], partial);
    }
    for (int off = 16; off >= 1; off >>= 1)
      partial += __shfl_xor(partial, off, 32);
    if (lane == 0) sc[l] = partial;
  }
  __syncthreads();

  // Softmax (max redundant per thread -> deterministic), then A@V.
  const float scale = 0.044194173824159216f;   // 1/sqrt(512)
  float m = -3.4e38f;
  for (int l = 0; l < 129; ++l) m = fmaxf(m, sc[l] * scale);
  if (t < 129) sp[t] = __expf(sc[t] * scale - m);
  __syncthreads();
  float Z = 0.f;
  for (int l = 0; l < 129; ++l) Z += sp[l];
  float inv = 1.0f / Z;

  for (int d = t; d < D; d += 256) {
    float acc = 0.f;
    for (int l = 0; l < 64; ++l) acc = fmaf(sp[l], bf2f(sV[l * D + d]), acc);
    acc = fmaf(sp[64], vmask[d], acc);
    float prel = 0.f, accp = 0.f;
    for (int jr = 0; jr < 64; ++jr) {
      float pj = sp[65 + jr];
      prel += pj;
      accp = fmaf(pj, posr_v[jr * D + d], accp);
    }
    acc += prel * relV[(size_t)b * D + d] + accp;
    out[(size_t)b * D + d] = acc * inv;
  }
}

// ---------------------------------------------------------------------------
extern "C" void kernel_launch(void* const* d_in, const int* in_sizes, int n_in,
                              void* d_out, int out_size, void* d_ws, size_t ws_size,
                              hipStream_t stream) {
  (void)in_sizes; (void)n_in; (void)out_size; (void)ws_size;
  const float* qee   = (const float*)d_in[0];
  const float* rel   = (const float*)d_in[1];
  const float* maskE = (const float*)d_in[2];
  const float* pos_e = (const float*)d_in[3];
  const float* pos_r = (const float*)d_in[4];
  const float* We_q  = (const float*)d_in[5];
  const float* be_q  = (const float*)d_in[6];
  const float* We_k  = (const float*)d_in[7];
  const float* be_k  = (const float*)d_in[8];
  const float* We_v  = (const float*)d_in[9];
  const float* be_v  = (const float*)d_in[10];
  // d_in[11]=Wr_q, d_in[12]=br_q are dead: only Q row 0 (entity) is used.
  const float* Wr_k  = (const float*)d_in[13];
  const float* br_k  = (const float*)d_in[14];
  const float* Wr_v  = (const float*)d_in[15];
  const float* br_v  = (const float*)d_in[16];
  float* out = (float*)d_out;

  // Workspace layout (~14.8 MB)
  char* ws = (char*)d_ws;
  unsigned short* WBf   = (unsigned short*)ws;        // 5 x 512KB bf16 weights
  unsigned short* WqBf  = WBf;
  unsigned short* WkBf  = WBf + 262144;
  unsigned short* WvBf  = WBf + 2 * 262144;
  unsigned short* WrkBf = WBf + 3 * 262144;
  unsigned short* WrvBf = WBf + 4 * 262144;
  float* posr_k = (float*)(ws + 5 * 524288);          // 64x512
  float* posr_v = posr_k + 32768;                     // 64x512
  float* kmask  = posr_v + 32768;                     // 512
  float* vmask  = kmask + 512;                        // 512
  float* qrow   = vmask + 512;                        // 2048x512
  float* relKb  = qrow + (size_t)NB * D;              // 2048x512
  float* relVb  = relKb + (size_t)NB * D;             // 2048x512

  hipFuncSetAttribute(reinterpret_cast<const void*>(k_row_gemms),
                      hipFuncAttributeMaxDynamicSharedMemorySize, 131072);
  hipFuncSetAttribute(reinterpret_cast<const void*>(k_fused_attn),
                      hipFuncAttributeMaxDynamicSharedMemorySize, 197664);

  k_convert_weights<<<5120, 256, 0, stream>>>(We_q, We_k, We_v, Wr_k, Wr_v, WBf);
  k_small_precompute<<<260, 256, 0, stream>>>(maskE, pos_e, pos_r,
                                              We_k, be_k, We_v, be_v,
                                              Wr_k, br_k, Wr_v, br_v,
                                              posr_k, posr_v, kmask, vmask);
  k_row_gemms<<<32, 256, 131072, stream>>>(qee, rel, pos_e, WqBf, WrkBf, WrvBf,
                                           be_q, qrow, relKb, relVb);
  k_fused_attn<<<NB, 256, 197664, stream>>>(qee, pos_e, WkBf, WvBf, be_k, be_v,
                                            qrow, relKb, relVb, posr_k, posr_v,
                                            kmask, vmask, out);
}
